// GVADE_For_Pretrain_75333726371974
// MI455X (gfx1250) — compile-verified, matrix operand order checked by reading, not run on hardware
//
#include <hip/hip_runtime.h>
#include <hip/hip_bf16.h>

typedef __attribute__((ext_vector_type(16))) __bf16 v16bf;
typedef __attribute__((ext_vector_type(8)))  float  v8f;

#define NEG_SLOPE 0.2f

union frag_cast { v16bf v; uint4 u[2]; };

// ---------------------------------------------------------------------------
// A fragment (16x32 bf16) gathered from row-major fp32, vectorized:
// lane L -> row (L&15); K pattern {h*8..h*8+7, 16+h*8..+7}, h=L>>4.
// All four 4-float runs are 16B aligned (lda, k0 multiples of 32).
// ---------------------------------------------------------------------------
__device__ inline v16bf frag_a_f32(const float* __restrict__ A, int lda,
                                   int row0, int k0, int lane) {
  const int m = lane & 15, half = lane >> 4;
  const float* p = A + (size_t)(row0 + m) * lda + k0 + half * 8;
  const float4 x0 = *(const float4*)(p);
  const float4 x1 = *(const float4*)(p + 4);
  const float4 x2 = *(const float4*)(p + 16);
  const float4 x3 = *(const float4*)(p + 20);
  v16bf f;
  f[0]  = (__bf16)x0.x; f[1]  = (__bf16)x0.y; f[2]  = (__bf16)x0.z; f[3]  = (__bf16)x0.w;
  f[4]  = (__bf16)x1.x; f[5]  = (__bf16)x1.y; f[6]  = (__bf16)x1.z; f[7]  = (__bf16)x1.w;
  f[8]  = (__bf16)x2.x; f[9]  = (__bf16)x2.y; f[10] = (__bf16)x2.z; f[11] = (__bf16)x2.w;
  f[12] = (__bf16)x3.x; f[13] = (__bf16)x3.y; f[14] = (__bf16)x3.z; f[15] = (__bf16)x3.w;
  return f;
}

// Fragment from LDS-resident row-major bf16 (ld = 64): two ds_load_b128
__device__ inline v16bf frag_lds(const __bf16* base, int row0, int k0, int lane) {
  const int m = lane & 15, half = lane >> 4;
  const __bf16* p = base + (row0 + m) * 64 + k0 + half * 8;
  frag_cast r;
  r.u[0] = *(const uint4*)p;
  r.u[1] = *(const uint4*)(p + 16);
  return r.v;
}

// ---------------------------------------------------------------------------
// Degree / normalization
// ---------------------------------------------------------------------------
__global__ void k_init_deg(float* deg, int n) {
  int i = blockIdx.x * blockDim.x + threadIdx.x;
  if (i < n) deg[i] = 1.0f;  // self-loop
}
__global__ void k_accum_deg(float* deg, const int* __restrict__ dst, int e) {
  int i = blockIdx.x * blockDim.x + threadIdx.x;
  if (i < e) atomicAdd(&deg[dst[i]], 1.0f);
}
__global__ void k_finalize_dinv(float* deg, int n) {
  int i = blockIdx.x * blockDim.x + threadIdx.x;
  if (i < n) deg[i] = rsqrtf(deg[i]);  // deg >= 1 always
}

// ---------------------------------------------------------------------------
// Weight pre-pack: W[K,N] fp32 -> bf16, fragment-major.
// Layout: tile (ct, kt) -> 32 lanes x 16 bf16 contiguous per lane:
//   Wp[(((ct*(K/32)) + kt)*32 + lane)*16 + j]
// ---------------------------------------------------------------------------
__global__ void k_pack_w(const float* __restrict__ W, __bf16* __restrict__ Wp,
                         int N, int K) {
  const int idx = blockIdx.x * blockDim.x + threadIdx.x;
  const int total = (N / 16) * (K / 32) * 32;
  if (idx >= total) return;
  const int lane = idx & 31;
  const int tile = idx >> 5;
  const int kt = tile % (K / 32);
  const int ct = tile / (K / 32);
  const int n = lane & 15, half = lane >> 4;
  const float* p = W + (size_t)(kt * 32 + half * 8) * N + ct * 16 + n;
  __bf16* q = Wp + (size_t)idx * 16;
#pragma unroll
  for (int j = 0; j < 8; ++j) q[j] = (__bf16)p[(size_t)j * N];
#pragma unroll
  for (int j = 0; j < 8; ++j) q[8 + j] = (__bf16)p[(size_t)(16 + j) * N];
}

// ---------------------------------------------------------------------------
// Dense projection: C[M,N] = A[M,K] * B[K,N], A fp32, B pre-packed bf16.
// block = 128 threads (4 waves); wave w -> 16x16 tile at
// (blockIdx.y*64 + w*16, blockIdx.x*16). M%64==0, N%16==0, K%32==0.
// ---------------------------------------------------------------------------
__global__ void k_gemm_wmma(const float* __restrict__ A, const __bf16* __restrict__ Bp,
                            float* __restrict__ C, int M, int N, int K) {
  const int lane = threadIdx.x & 31;
  const int wave = threadIdx.x >> 5;
  const int row0 = (blockIdx.y * 4 + wave) * 16;
  const int ct   = blockIdx.x;                 // column tile
  const int kt_n = K / 32;
  const __bf16* bbase = Bp + ((size_t)ct * kt_n * 32 + lane) * 16;
  v8f acc = {};
  for (int kt = 0; kt < kt_n; ++kt) {
    v16bf a = frag_a_f32(A, K, row0, kt * 32, lane);
    frag_cast b;
    const uint4* bp = (const uint4*)(bbase + (size_t)kt * 32 * 16);
    b.u[0] = bp[0];
    b.u[1] = bp[1];
    acc = __builtin_amdgcn_wmma_f32_16x16x32_bf16(false, a, false, b.v,
                                                  (short)0, acc, false, false);
  }
  const int half = lane >> 4, n = lane & 15;
  float* crow = C + (size_t)(row0 + half * 8) * N + ct * 16 + n;
#pragma unroll
  for (int r = 0; r < 8; ++r) crow[(size_t)r * N] = acc[r];
}

// ---------------------------------------------------------------------------
// Aggregation init: out[i] = h[i]*dinv[i]^2 + bias   (self-loop + bias), float4
// ---------------------------------------------------------------------------
__global__ void k_agg_init(const float4* __restrict__ h4, const float* __restrict__ dinv,
                           const float4* __restrict__ bias4, float4* __restrict__ out4,
                           int n, int F) {
  const int C = F >> 2;
  int idx = blockIdx.x * blockDim.x + threadIdx.x;
  if (idx >= n * C) return;
  int i = idx / C, c = idx - i * C;
  float dv = dinv[i];
  float s = dv * dv;
  float4 v = h4[idx];
  float4 b = bias4[c];
  out4[idx] = make_float4(v.x * s + b.x, v.y * s + b.y, v.z * s + b.z, v.w * s + b.w);
}

// Edge scatter: out[d] += h[s] * dinv[s]*dinv[d]   (float4 gather + 4 atomics)
__global__ void k_agg_scatter(const float* __restrict__ h, const int* __restrict__ src,
                              const int* __restrict__ dst, const float* __restrict__ dinv,
                              float* __restrict__ out, int E, int F) {
  const int C = F >> 2;
  long long idx = (long long)blockIdx.x * blockDim.x + threadIdx.x;
  if (idx >= (long long)E * C) return;
  int e = (int)(idx / C);
  int c = (int)(idx - (long long)e * C);
  int s = src[e], d = dst[e];
  float nrm = dinv[s] * dinv[d];
  const float4 v = ((const float4*)(h + (size_t)s * F))[c];
  float* o = out + (size_t)d * F + (size_t)c * 4;
  atomicAdd(o + 0, v.x * nrm);
  atomicAdd(o + 1, v.y * nrm);
  atomicAdd(o + 2, v.z * nrm);
  atomicAdd(o + 3, v.w * nrm);
}

__global__ void k_leaky(float4* __restrict__ z, int n4) {
  int i = blockIdx.x * blockDim.x + threadIdx.x;
  if (i >= n4) return;
  float4 v = z[i];
  v.x = v.x > 0.0f ? v.x : NEG_SLOPE * v.x;
  v.y = v.y > 0.0f ? v.y : NEG_SLOPE * v.y;
  v.z = v.z > 0.0f ? v.z : NEG_SLOPE * v.z;
  v.w = v.w > 0.0f ? v.w : NEG_SLOPE * v.w;
  z[i] = v;
}

__global__ void k_cvt_bf16(const float* __restrict__ z, __bf16* __restrict__ zb, int n) {
  int i = blockIdx.x * blockDim.x + threadIdx.x;
  if (i < n) zb[i] = (__bf16)z[i];
}

// ---------------------------------------------------------------------------
// Decoder: Out = sigmoid(Z Z^T), Z bf16 [N,64].
// Block = 256 thr (8 waves) -> 128x128 output tile. Stage the block's A-rows
// and B-rows of Z in LDS (2 x 16KB), then per wave: hold A frags, sweep 8
// column tiles with 2 WMMAs each (K=64), fuse sigmoid into the 1 GiB store.
// ---------------------------------------------------------------------------
__global__ void k_decoder_wmma(const __bf16* __restrict__ Zb, float* __restrict__ Out,
                               int N) {
  __shared__ alignas(16) __bf16 lA[128 * 64];
  __shared__ alignas(16) __bf16 lB[128 * 64];
  const int tid = threadIdx.x;
  const int lane = tid & 31, wave = tid >> 5;
  const int rowb = blockIdx.y * 128;
  const int colb = blockIdx.x * 128;

  // cooperative staging: 2 x 1024 uint4, 256 threads -> 4 each per buffer
  const uint4* gA = (const uint4*)(Zb + (size_t)rowb * 64);
  const uint4* gB = (const uint4*)(Zb + (size_t)colb * 64);
  uint4* sA = (uint4*)lA;
  uint4* sB = (uint4*)lB;
#pragma unroll
  for (int i = 0; i < 4; ++i) {
    sA[tid + 256 * i] = gA[tid + 256 * i];
    sB[tid + 256 * i] = gB[tid + 256 * i];
  }
  __syncthreads();

  const v16bf a0 = frag_lds(lA, wave * 16, 0, lane);
  const v16bf a1 = frag_lds(lA, wave * 16, 32, lane);
  const int half = lane >> 4, n = lane & 15;
  const int row0 = rowb + wave * 16;
#pragma unroll
  for (int jt = 0; jt < 8; ++jt) {
    v16bf b0 = frag_lds(lB, jt * 16, 0, lane);   // B[k][n] = Z[n][k]
    v16bf b1 = frag_lds(lB, jt * 16, 32, lane);
    v8f acc = {};
    acc = __builtin_amdgcn_wmma_f32_16x16x32_bf16(false, a0, false, b0,
                                                  (short)0, acc, false, false);
    acc = __builtin_amdgcn_wmma_f32_16x16x32_bf16(false, a1, false, b1,
                                                  (short)0, acc, false, false);
    float* orow = Out + (size_t)(row0 + half * 8) * N + colb + jt * 16 + n;
#pragma unroll
    for (int r = 0; r < 8; ++r)
      orow[(size_t)r * N] = 1.0f / (1.0f + __expf(-acc[r]));
  }
}

// ---------------------------------------------------------------------------
extern "C" void kernel_launch(void* const* d_in, const int* in_sizes, int n_in,
                              void* d_out, int out_size, void* d_ws, size_t ws_size,
                              hipStream_t stream) {
  const float* X   = (const float*)d_in[0];
  const int*   ei  = (const int*)d_in[1];
  const float* W1  = (const float*)d_in[2];
  const float* b1  = (const float*)d_in[3];
  const float* W2  = (const float*)d_in[4];
  const float* b2  = (const float*)d_in[5];
  const float* Wmu = (const float*)d_in[6];
  const float* bmu = (const float*)d_in[7];
  float* out = (float*)d_out;

  const int IN = 512, H1 = 256, H2 = 128, L = 64;
  const int N = in_sizes[0] / IN;   // 16384
  const int E = in_sizes[1] / 2;    // 524288
  const int* src = ei;
  const int* dstv = ei + E;

  char* ws = (char*)d_ws;
  float*  dinv = (float*)ws;  ws += (size_t)N * sizeof(float);
  float*  h    = (float*)ws;  ws += (size_t)N * 256 * sizeof(float);
  float*  bufA = (float*)ws;  ws += (size_t)N * 256 * sizeof(float);  // L1 / L3 out
  float*  bufB = (float*)ws;  ws += (size_t)N * 128 * sizeof(float);  // L2 out
  __bf16* zb   = (__bf16*)ws; ws += (size_t)N * L * sizeof(__bf16);
  __bf16* W1p  = (__bf16*)ws; ws += (size_t)IN * H1 * sizeof(__bf16);
  __bf16* W2p  = (__bf16*)ws; ws += (size_t)H1 * H2 * sizeof(__bf16);
  __bf16* Wmp  = (__bf16*)ws;

  const int T = 256;
  // symmetric normalization
  k_init_deg<<<(N + T - 1) / T, T, 0, stream>>>(dinv, N);
  k_accum_deg<<<(E + T - 1) / T, T, 0, stream>>>(dinv, dstv, E);
  k_finalize_dinv<<<(N + T - 1) / T, T, 0, stream>>>(dinv, N);

  // weight pre-pack (fragment-major bf16)
  k_pack_w<<<((H1 / 16) * (IN / 32) * 32 + T - 1) / T, T, 0, stream>>>(W1, W1p, H1, IN);
  k_pack_w<<<((H2 / 16) * (H1 / 32) * 32 + T - 1) / T, T, 0, stream>>>(W2, W2p, H2, H1);
  k_pack_w<<<((L  / 16) * (H2 / 32) * 32 + T - 1) / T, T, 0, stream>>>(Wmu, Wmp, L, H2);

  // ---- layer 1: h = X@W1 ; aggregate ; leaky ----
  k_gemm_wmma<<<dim3(H1 / 16, N / 64), 128, 0, stream>>>(X, W1p, h, N, H1, IN);
  k_agg_init<<<(N * (H1 / 4) + T - 1) / T, T, 0, stream>>>(
      (const float4*)h, dinv, (const float4*)b1, (float4*)bufA, N, H1);
  k_agg_scatter<<<(unsigned)(((long long)E * (H1 / 4) + T - 1) / T), T, 0, stream>>>(
      h, src, dstv, dinv, bufA, E, H1);
  k_leaky<<<(N * (H1 / 4) + T - 1) / T, T, 0, stream>>>((float4*)bufA, N * (H1 / 4));

  // ---- layer 2 ----
  k_gemm_wmma<<<dim3(H2 / 16, N / 64), 128, 0, stream>>>(bufA, W2p, h, N, H2, H1);
  k_agg_init<<<(N * (H2 / 4) + T - 1) / T, T, 0, stream>>>(
      (const float4*)h, dinv, (const float4*)b2, (float4*)bufB, N, H2);
  k_agg_scatter<<<(unsigned)(((long long)E * (H2 / 4) + T - 1) / T), T, 0, stream>>>(
      h, src, dstv, dinv, bufB, E, H2);
  k_leaky<<<(N * (H2 / 4) + T - 1) / T, T, 0, stream>>>((float4*)bufB, N * (H2 / 4));

  // ---- layer 3 (mu, no activation) ----
  k_gemm_wmma<<<dim3(L / 16, N / 64), 128, 0, stream>>>(bufB, Wmp, h, N, L, H2);
  k_agg_init<<<(N * (L / 4) + T - 1) / T, T, 0, stream>>>(
      (const float4*)h, dinv, (const float4*)bmu, (float4*)bufA, N, L);
  k_agg_scatter<<<(unsigned)(((long long)E * (L / 4) + T - 1) / T), T, 0, stream>>>(
      h, src, dstv, dinv, bufA, E, L);

  // ---- decoder: sigmoid(Z Z^T) ----
  k_cvt_bf16<<<(N * L + T - 1) / T, T, 0, stream>>>(bufA, zb, N * L);
  k_decoder_wmma<<<dim3(N / 128, N / 128), 256, 0, stream>>>(zb, out, N);
}
